// DeepSet_14164802142742
// MI455X (gfx1250) — matrix operand order repaired and checked by reading, not compiled, not run on hardware
//
#include <hip/hip_runtime.h>
#include <hip/hip_bf16.h>

typedef __attribute__((ext_vector_type(16))) __bf16 v16bf;
typedef __attribute__((ext_vector_type(8)))  __bf16 v8bf;
typedef __attribute__((ext_vector_type(8)))  float  v8f;

#define NODE   64
#define HID    128
#define OUTD   16
#define NSEG   1024
#define NT_HID 8      // 128 / 16 column tiles
#define KC_NODE 2     // 64  / 32 k-chunks
#define KC_HID  4     // 128 / 32 k-chunks

#define W1_FRAG_ELEMS (KC_NODE * NT_HID * 512)   // 8192 bf16
#define W2_FRAG_ELEMS (KC_HID  * NT_HID * 512)   // 16384 bf16
#define WFRAG_ELEMS   (W1_FRAG_ELEMS + W2_FRAG_ELEMS) // 24576 bf16 = 48 KB

static __device__ __forceinline__ void atomic_add_f32(float* p, float v) {
  __hip_atomic_fetch_add(p, v, __ATOMIC_RELAXED, __HIP_MEMORY_SCOPE_AGENT);
}

// ---------------------------------------------------------------------------
// Pack a row-major f32 weight [K x Ncols] into per-lane bf16 WMMA B-fragments.
// Fragment f = kc*NT + nt covers K rows [kc*32, kc*32+32), cols [nt*16, nt*16+16).
// B layout (16-bit 32x16): lane = h*16 + c holds column c, K = h*8 + e (e<8), 16 + h*8 + (e-8) (e>=8),
// stored contiguously: dst[f*512 + lane*16 + e].
// ---------------------------------------------------------------------------
__global__ void pack_w_kernel(const float* __restrict__ W, __bf16* __restrict__ dst,
                              int Ncols, int NT) {
  const int f    = blockIdx.x;       // kc*NT + nt
  const int lane = threadIdx.x;      // 0..31
  const int kc = f / NT, nt = f % NT;
  const int h = lane >> 4, c = lane & 15;
  const int col = nt * 16 + c;
  #pragma unroll
  for (int e = 0; e < 16; ++e) {
    const int k = kc * 32 + (e < 8 ? h * 8 + e : 16 + h * 8 + (e - 8));
    dst[(size_t)f * 512 + lane * 16 + e] = (__bf16)W[(size_t)k * Ncols + col];
  }
}

__global__ void zero_ws_kernel(float* __restrict__ p, int n) {
  int i = blockIdx.x * blockDim.x + threadIdx.x;
  if (i < n) p[i] = 0.0f;
}

// ---------------------------------------------------------------------------
// phi + segment-sum scatter. One block = 256 threads = 8 wave32; each wave owns
// a 16-row node tile. h2 = relu(x@W1+b1)@W2+b2, then reduced into sums/counts.
// W1/W2 fragments are staged into LDS once per block; B operands come from DS.
// Requires N % 128 == 0 (true for the harness N = 2,000,000).
// ---------------------------------------------------------------------------
__global__ __launch_bounds__(256)
void phi_scatter_kernel(const float* __restrict__ x, const int* __restrict__ xb,
                        const __bf16* __restrict__ pW12,   // W1 frags ++ W2 frags
                        const float* __restrict__ b1, const float* __restrict__ b2,
                        float* __restrict__ sums, float* __restrict__ counts, int N) {
  __shared__ __align__(16) __bf16 lw[WFRAG_ELEMS];    // 48 KB weight fragments
  __shared__ __align__(16) __bf16 hl[8 * 16 * HID];   // 32 KB: per-wave 16x128 bf16
  if (blockIdx.x * 128 >= N) return;                  // block-uniform guard

  const int lane = threadIdx.x & 31;
  const int w    = threadIdx.x >> 5;
  const int h    = lane >> 4;          // half-wave
  const int c    = lane & 15;          // col in C-tile / row in A-frag
  const int tileRow = blockIdx.x * 128 + w * 16;
  __bf16* hw  = hl + w * 16 * HID;
  const __bf16* lW1 = lw;
  const __bf16* lW2 = lw + W1_FRAG_ELEMS;

  // Prefetch next block's x tile (streaming read sets the roofline floor).
  {
    const size_t nextBase = (size_t)(blockIdx.x + 1) * 128 * NODE;
    if (nextBase < (size_t)N * NODE)
      __builtin_prefetch(x + nextBase + (size_t)threadIdx.x * 32, 0, 1);
  }

  // ---- cooperative stage of W1+W2 fragments into LDS (48 KB, v128 chunks) ----
  {
    const v8bf* src = (const v8bf*)pW12;
    v8bf*       dst = (v8bf*)lw;
    #pragma unroll
    for (int i = 0; i < WFRAG_ELEMS / 8 / 256; ++i)      // 12 iterations
      dst[threadIdx.x + i * 256] = src[threadIdx.x + i * 256];
  }
  __syncthreads();

  // ---- GEMM1: 16x128 = (16x64) @ (64x128), K in 2 chunks of 32 ----
  v8f acc[NT_HID];
  #pragma unroll
  for (int nt = 0; nt < NT_HID; ++nt) acc[nt] = (v8f){};

  #pragma unroll
  for (int kc = 0; kc < KC_NODE; ++kc) {
    // A fragment: lane holds row (tileRow + c), K = kc*32 + h*8 + {0..7, 16..23}
    const float* xr = x + (size_t)(tileRow + c) * NODE + kc * 32 + h * 8;
    v16bf a;
    #pragma unroll
    for (int e = 0; e < 8; ++e) {
      a[e]     = (__bf16)xr[e];
      a[e + 8] = (__bf16)xr[16 + e];
    }
    #pragma unroll
    for (int nt = 0; nt < NT_HID; ++nt) {
      v16bf b = *(const v16bf*)(lW1 + ((kc * NT_HID + nt) * 32 + lane) * 16);
      acc[nt] = __builtin_amdgcn_wmma_f32_16x16x32_bf16(false, a, false, b,
                                                        (short)0, acc[nt], false, false);
    }
  }

  // bias + ReLU, spill to LDS as bf16 row-major 16x128 (A-frag source layout)
  #pragma unroll
  for (int nt = 0; nt < NT_HID; ++nt) {
    const float bv = b1[nt * 16 + c];
    #pragma unroll
    for (int v = 0; v < 8; ++v) {
      float t = acc[nt][v] + bv;           // C layout: row v+8h, col nt*16+c
      t = t > 0.0f ? t : 0.0f;
      hw[(v + 8 * h) * HID + nt * 16 + c] = (__bf16)t;
    }
  }
  __syncthreads();

  // ---- GEMM2: 16x128 = (16x128) @ (128x128), K in 4 chunks of 32 ----
  v8f acc2[NT_HID];
  #pragma unroll
  for (int nt = 0; nt < NT_HID; ++nt) acc2[nt] = (v8f){};

  #pragma unroll
  for (int kc = 0; kc < KC_HID; ++kc) {
    const __bf16* hr = hw + c * HID + kc * 32 + h * 8;
    v8bf lo = *(const v8bf*)hr;
    v8bf hi = *(const v8bf*)(hr + 16);
    v16bf a;
    #pragma unroll
    for (int e = 0; e < 8; ++e) { a[e] = lo[e]; a[e + 8] = hi[e]; }
    #pragma unroll
    for (int nt = 0; nt < NT_HID; ++nt) {
      v16bf b = *(const v16bf*)(lW2 + ((kc * NT_HID + nt) * 32 + lane) * 16);
      acc2[nt] = __builtin_amdgcn_wmma_f32_16x16x32_bf16(false, a, false, b,
                                                         (short)0, acc2[nt], false, false);
    }
  }
  // + b2 (no ReLU on second layer)
  #pragma unroll
  for (int nt = 0; nt < NT_HID; ++nt) {
    const float bv = b2[nt * 16 + c];
    #pragma unroll
    for (int v = 0; v < 8; ++v) acc2[nt][v] += bv;
  }

  // ---- sorted segment scatter ----
  const int sid  = xb[tileRow + c];
  const int sid0 = __shfl(sid, 0, 32);
  const bool uniform = __all(sid == sid0);

  if (uniform) {
    // whole 16-row tile belongs to one segment: reduce rows in-register
    #pragma unroll
    for (int nt = 0; nt < NT_HID; ++nt) {
      float s = 0.0f;
      #pragma unroll
      for (int v = 0; v < 8; ++v) s += acc2[nt][v];   // rows {0..7}+8h at col c
      s += __shfl_xor(s, 16, 32);                      // combine both half-waves
      if (h == 0)
        atomic_add_f32(&sums[(size_t)sid0 * HID + nt * 16 + c], s);
    }
    if (lane == 0) atomic_add_f32(&counts[sid0], 16.0f);
  } else {
    // segment boundary inside tile (rare): per-row atomics
    #pragma unroll
    for (int v = 0; v < 8; ++v) {
      const int sr = xb[tileRow + v + 8 * h];
      #pragma unroll
      for (int nt = 0; nt < NT_HID; ++nt)
        atomic_add_f32(&sums[(size_t)sr * HID + nt * 16 + c], acc2[nt][v]);
    }
    if (h == 0) atomic_add_f32(&counts[sid], 1.0f);
  }
}

// ---------------------------------------------------------------------------
// rho: out = relu((sums/max(counts,1)) @ W3 + b3) @ W4 + b4.  1024 segments ->
// 64 row tiles; 8 blocks x 8 waves, one tile per wave. Tiny; weights from L2.
// ---------------------------------------------------------------------------
__global__ __launch_bounds__(256)
void rho_kernel(const float* __restrict__ sums, const float* __restrict__ counts,
                const __bf16* __restrict__ pW3, const float* __restrict__ b3,
                const __bf16* __restrict__ pW4, const float* __restrict__ b4,
                float* __restrict__ out) {
  __shared__ __align__(16) __bf16 hl[8 * 16 * HID];
  const int lane = threadIdx.x & 31;
  const int w    = threadIdx.x >> 5;
  const int h    = lane >> 4;
  const int c    = lane & 15;
  const int t    = blockIdx.x * 8 + w;     // segment tile 0..63
  const int r    = t * 16 + c;             // this lane's segment row
  __bf16* hw = hl + w * 16 * HID;

  const float inv = 1.0f / fmaxf(counts[r], 1.0f);

  // ---- GEMM3: (16x128) @ (128x128) ----
  v8f acc[NT_HID];
  #pragma unroll
  for (int nt = 0; nt < NT_HID; ++nt) acc[nt] = (v8f){};

  #pragma unroll
  for (int kc = 0; kc < KC_HID; ++kc) {
    const float* sr = sums + (size_t)r * HID + kc * 32 + h * 8;
    v16bf a;
    #pragma unroll
    for (int e = 0; e < 8; ++e) {
      a[e]     = (__bf16)(sr[e] * inv);
      a[e + 8] = (__bf16)(sr[16 + e] * inv);
    }
    #pragma unroll
    for (int nt = 0; nt < NT_HID; ++nt) {
      v16bf b = *(const v16bf*)(pW3 + ((size_t)(kc * NT_HID + nt) * 32 + lane) * 16);
      acc[nt] = __builtin_amdgcn_wmma_f32_16x16x32_bf16(false, a, false, b,
                                                        (short)0, acc[nt], false, false);
    }
  }
  #pragma unroll
  for (int nt = 0; nt < NT_HID; ++nt) {
    const float bv = b3[nt * 16 + c];
    #pragma unroll
    for (int v = 0; v < 8; ++v) {
      float tv = acc[nt][v] + bv;
      tv = tv > 0.0f ? tv : 0.0f;
      hw[(v + 8 * h) * HID + nt * 16 + c] = (__bf16)tv;
    }
  }
  __syncthreads();

  // ---- GEMM4: (16x128) @ (128x16) ----
  v8f accO = (v8f){};
  #pragma unroll
  for (int kc = 0; kc < KC_HID; ++kc) {
    const __bf16* hr = hw + c * HID + kc * 32 + h * 8;
    v8bf lo = *(const v8bf*)hr;
    v8bf hi = *(const v8bf*)(hr + 16);
    v16bf a;
    #pragma unroll
    for (int e = 0; e < 8; ++e) { a[e] = lo[e]; a[e + 8] = hi[e]; }
    v16bf b = *(const v16bf*)(pW4 + ((size_t)kc * 32 + lane) * 16);   // NT=1
    accO = __builtin_amdgcn_wmma_f32_16x16x32_bf16(false, a, false, b,
                                                   (short)0, accO, false, false);
  }
  const float bo = b4[c];
  #pragma unroll
  for (int v = 0; v < 8; ++v)
    out[(size_t)(t * 16 + v + 8 * h) * OUTD + c] = accO[v] + bo;
}

// ---------------------------------------------------------------------------
extern "C" void kernel_launch(void* const* d_in, const int* in_sizes, int n_in,
                              void* d_out, int out_size, void* d_ws, size_t ws_size,
                              hipStream_t stream) {
  const float* x  = (const float*)d_in[0];
  const int*   xb = (const int*)  d_in[1];
  const float* W1 = (const float*)d_in[2];
  const float* b1 = (const float*)d_in[3];
  const float* W2 = (const float*)d_in[4];
  const float* b2 = (const float*)d_in[5];
  const float* W3 = (const float*)d_in[6];
  const float* b3 = (const float*)d_in[7];
  const float* W4 = (const float*)d_in[8];
  const float* b4 = (const float*)d_in[9];
  float* out = (float*)d_out;
  const int N = in_sizes[0] / NODE;

  // workspace layout (all offsets 256B-aligned); pW1 and pW2 are contiguous so
  // the phi kernel can stage both with one LDS copy.
  char* ws = (char*)d_ws;
  float*  sums   = (float*)(ws);                 // 1024*128 f32 = 512 KB
  float*  counts = (float*)(ws + 524288);        // 1024 f32 (contiguous with sums)
  __bf16* pW1    = (__bf16*)(ws + 528384);       // 16 frags * 512 bf16 = 16 KB
  __bf16* pW2    = (__bf16*)(ws + 544768);       // 32 frags           = 32 KB
  __bf16* pW3    = (__bf16*)(ws + 577536);       // 32 frags           = 32 KB
  __bf16* pW4    = (__bf16*)(ws + 610304);       // 4 frags            =  4 KB

  const int nzero = NSEG * HID + NSEG;           // sums + counts (contiguous)
  zero_ws_kernel<<<(nzero + 255) / 256, 256, 0, stream>>>(sums, nzero);

  pack_w_kernel<<<KC_NODE * NT_HID, 32, 0, stream>>>(W1, pW1, HID,  NT_HID);
  pack_w_kernel<<<KC_HID  * NT_HID, 32, 0, stream>>>(W2, pW2, HID,  NT_HID);
  pack_w_kernel<<<KC_HID  * NT_HID, 32, 0, stream>>>(W3, pW3, HID,  NT_HID);
  pack_w_kernel<<<KC_HID  * 1,      32, 0, stream>>>(W4, pW4, OUTD, 1);

  phi_scatter_kernel<<<(N + 127) / 128, 256, 0, stream>>>(x, xb, pW1, b1, b2,
                                                          sums, counts, N);
  rho_kernel<<<8, 256, 0, stream>>>(sums, counts, pW3, b3, pW4, b4, out);
}
